// _UnimolDecoderLayer_38070590112234
// MI455X (gfx1250) — compile-verified, matrix operand order checked by reading, not run on hardware
//
#include <hip/hip_runtime.h>
#include <hip/hip_bf16.h>
#include <math.h>

// Problem dims (compile-time; match reference)
#define LQ   1024
#define BQ   4
#define DQ   1024
#define HQ   16
#define HDQ  64
#define DFFQ 4096
#define TOK  4096   // L*B tokens
#define ZB   64     // B*H attention batches
#define EPSQ 1e-5f

typedef __attribute__((ext_vector_type(16))) _Float16 v16h;
typedef __attribute__((ext_vector_type(8)))  float    v8f;

union FragH { v16h v; uint4 q[2]; };

__device__ __forceinline__ v8f v8f_zero() {
  v8f z = {0.f,0.f,0.f,0.f,0.f,0.f,0.f,0.f};
  return z;
}

// CDNA5 async global->LDS DMA (VGLOBAL encoding, tracked by ASYNCcnt).
// vdst VGPR = LDS byte address; per ISA aperture rule the low 32 bits of a
// generic pointer into LDS ARE the LDS byte address, so the cast is exact.
__device__ __forceinline__ void async_g2l_b128(void* lds_ptr, const void* gptr) {
  unsigned int laddr = (unsigned int)(unsigned long long)lds_ptr;
  asm volatile("global_load_async_to_lds_b128 %0, %1, off"
               :: "v"(laddr), "v"(gptr)
               : "memory");
}

__device__ __forceinline__ void wait_async0() {
  asm volatile("s_wait_asynccnt 0x0" ::: "memory");
}

// ---------------------------------------------------------------------------
// WMMA GEMM core: C[M,N] = A[M,K] * W[N,K]^T (both row-major, K contiguous).
// Block = 256 threads = 8 wave32. Wave tile = 64(M) x 32(N) via 4x2 WMMA
// 16x16 accumulators. Block tile = (WM*64) x (WN*32), WM*WN == 8.
// K advanced in steps of 32 (one v_wmma_f32_16x16x32_f16 per subtile pair).
// Operands double-buffered through LDS: staged by global_load_async_to_lds_b128
// (no VGPR round trip, overlapped with WMMA via ASYNCcnt), consumed with
// ds_load_b128 (pitch 40 halves = 80B rows -> 16B aligned, conflict-free).
//
// Fragment layouts per CDNA5 ISA 7.12.2 (16-bit A 16x32): lanes 0-15 hold
// rows, halves [lh*8 .. lh*8+8) then [16+lh*8 ..). B (K x N) for a row-major
// W[N,K] tile is loaded with the identical per-lane indexing (lane = N).
// C/D: VGPR r -> M = r + 8*(lane>=16), N = lane&15.
// ---------------------------------------------------------------------------
template<int WM, int WN, typename EP>
__device__ __forceinline__ void gemm_core(const _Float16* __restrict__ A, long lda,
                                          const _Float16* __restrict__ Bw, long ldb,
                                          int K, EP ep) {
  constexpr int BM = WM * 64;
  constexpr int BN = WN * 32;
  constexpr int P  = 40;                 // LDS row pitch in halves
  __shared__ ushort sm[2][(BM + BN) * P];

  const int tid  = threadIdx.x;
  const int wave = tid >> 5;
  const int lane = tid & 31;
  const int wm   = wave / WN;
  const int wn   = wave % WN;
  const int lr   = lane & 15;            // row within 16x16 fragment
  const int lh   = lane >> 4;            // K-half selector

  A  += (long)blockIdx.y * BM * lda;
  Bw += (long)blockIdx.x * BN * ldb;

  v8f acc[4][2];
#pragma unroll
  for (int i = 0; i < 4; ++i)
#pragma unroll
    for (int j = 0; j < 2; ++j) acc[i][j] = v8f_zero();

  // Cooperative async staging: each k-step DMAs BM*64B (A) + BN*64B (B)
  // straight into LDS; 4 x 16B chunks per 32-half row, 256 threads.
  auto stage = [&](int buf, int k0) {
    ushort* sA = sm[buf];
    ushort* sB = sm[buf] + BM * P;
#pragma unroll
    for (int i = 0; i < BM / 64; ++i) {
      int c = tid + i * 256;
      int r = c >> 2, col = (c & 3) * 8;
      async_g2l_b128(sA + r * P + col, A + (long)r * lda + k0 + col);
    }
#pragma unroll
    for (int i = 0; i < BN / 64; ++i) {
      int c = tid + i * 256;
      int r = c >> 2, col = (c & 3) * 8;
      async_g2l_b128(sB + r * P + col, Bw + (long)r * ldb + k0 + col);
    }
  };

  stage(0, 0);
  const int nk = K >> 5;
  for (int kk = 0; kk < nk; ++kk) {
    // Each wave drains its own in-flight async->LDS stores, then the
    // work-group barrier makes the freshly staged buffer globally visible.
    // (ds_loads of the previous buffer were already drained by the WMMA
    // consumers under s_wait_dscnt before reaching the barrier.)
    wait_async0();
    __syncthreads();
    const int cur = kk & 1;
    if (kk + 1 < nk) stage(cur ^ 1, (kk + 1) * 32);

    const ushort* sA = sm[cur];
    const ushort* sB = sm[cur] + BM * P;
    FragH a[4], b[2];
#pragma unroll
    for (int mi = 0; mi < 4; ++mi) {
      const ushort* p = sA + (wm * 64 + mi * 16 + lr) * P + lh * 8;
      a[mi].q[0] = *(const uint4*)p;
      a[mi].q[1] = *(const uint4*)(p + 16);
    }
#pragma unroll
    for (int ni = 0; ni < 2; ++ni) {
      const ushort* p = sB + (wn * 32 + ni * 16 + lr) * P + lh * 8;
      b[ni].q[0] = *(const uint4*)p;
      b[ni].q[1] = *(const uint4*)(p + 16);
    }
#pragma unroll
    for (int mi = 0; mi < 4; ++mi)
#pragma unroll
      for (int ni = 0; ni < 2; ++ni)
        acc[mi][ni] = __builtin_amdgcn_wmma_f32_16x16x32_f16(
            false, a[mi].v, false, b[ni].v, (short)0, acc[mi][ni], false, false);
  }

  // Epilogue in registers; per ISA C/D layout.
  const int m0 = (int)blockIdx.y * BM + wm * 64 + lh * 8;
  const int n0 = (int)blockIdx.x * BN + wn * 32 + lr;
#pragma unroll
  for (int mi = 0; mi < 4; ++mi)
#pragma unroll
    for (int ni = 0; ni < 2; ++ni)
#pragma unroll
      for (int r = 0; r < 8; ++r)
        ep(m0 + mi * 16 + r, n0 + ni * 16, acc[mi][ni][r]);
}

// ---------------------------------------------------------------------------
// Elementwise / reduction kernels
// ---------------------------------------------------------------------------

__global__ void k_f2h(const float* __restrict__ s, _Float16* __restrict__ d, int n) {
  int i = blockIdx.x * 256 + threadIdx.x;
  int j = i * 4;
  if (j < n) {
    float4 v = ((const float4*)s)[i];
    d[j + 0] = (_Float16)v.x;
    d[j + 1] = (_Float16)v.y;
    d[j + 2] = (_Float16)v.z;
    d[j + 3] = (_Float16)v.w;
  }
}

// LayerNorm over D=1024; one block (256 thr) per token row; f16 output for WMMA.
__global__ void __launch_bounds__(256) k_ln(const float* __restrict__ x,
                                            const float* __restrict__ w,
                                            const float* __restrict__ b,
                                            _Float16* __restrict__ o) {
  __shared__ float s1[256];
  __shared__ float s2[256];
  const int tid = threadIdx.x;
  const long row = blockIdx.x;
  float4 v = ((const float4*)(x + row * DQ))[tid];
  float sum = v.x + v.y + v.z + v.w;
  float sq  = v.x * v.x + v.y * v.y + v.z * v.z + v.w * v.w;
  s1[tid] = sum; s2[tid] = sq;
  __syncthreads();
#pragma unroll
  for (int s = 128; s > 0; s >>= 1) {
    if (tid < s) { s1[tid] += s1[tid + s]; s2[tid] += s2[tid + s]; }
    __syncthreads();
  }
  const float mu = s1[0] * (1.0f / DQ);
  const float var = s2[0] * (1.0f / DQ) - mu * mu;
  const float rs = rsqrtf(var + EPSQ);
  const int j = tid * 4;
  _Float16* op = o + row * DQ + j;
  op[0] = (_Float16)((v.x - mu) * rs * w[j + 0] + b[j + 0]);
  op[1] = (_Float16)((v.y - mu) * rs * w[j + 1] + b[j + 1]);
  op[2] = (_Float16)((v.z - mu) * rs * w[j + 2] + b[j + 2]);
  op[3] = (_Float16)((v.w - mu) * rs * w[j + 3] + b[j + 3]);
}

// Softmax over last dim (1024) of the logits (== edge_out), f16 output.
__global__ void __launch_bounds__(256) k_softmax(const float* __restrict__ t,
                                                 _Float16* __restrict__ a) {
  __shared__ float sb[256];
  const int tid = threadIdx.x;
  const long row = blockIdx.x;
  float4 v = ((const float4*)(t + row * LQ))[tid];
  float mx = fmaxf(fmaxf(v.x, v.y), fmaxf(v.z, v.w));
  sb[tid] = mx;
  __syncthreads();
#pragma unroll
  for (int s = 128; s > 0; s >>= 1) {
    if (tid < s) sb[tid] = fmaxf(sb[tid], sb[tid + s]);
    __syncthreads();
  }
  mx = sb[0];
  __syncthreads();
  float4 e;
  e.x = __expf(v.x - mx); e.y = __expf(v.y - mx);
  e.z = __expf(v.z - mx); e.w = __expf(v.w - mx);
  sb[tid] = e.x + e.y + e.z + e.w;
  __syncthreads();
#pragma unroll
  for (int s = 128; s > 0; s >>= 1) {
    if (tid < s) sb[tid] += sb[tid + s];
    __syncthreads();
  }
  const float inv = 1.0f / sb[0];
  _Float16* op = a + row * LQ + tid * 4;
  op[0] = (_Float16)(e.x * inv);
  op[1] = (_Float16)(e.y * inv);
  op[2] = (_Float16)(e.z * inv);
  op[3] = (_Float16)(e.w * inv);
}

// vt[z][d][m] = V[m, b, hh*64 + d]  (transpose so attn@V is an A@W^T GEMM)
__global__ void k_vtrans(const _Float16* __restrict__ qkv, _Float16* __restrict__ vt) {
  const int z = blockIdx.z, b = z >> 4, hh = z & 15;
  const int d = blockIdx.y;
  const int m = blockIdx.x * 256 + threadIdx.x;
  vt[(((long)z * HDQ) + d) * LQ + m] =
      qkv[((long)m * BQ + b) * 3072 + 2048 + hh * HDQ + d];
}

// ---------------------------------------------------------------------------
// GEMM kernels with fused epilogues
// ---------------------------------------------------------------------------

// qkv = ln1(x) @ in_proj_w^T + in_proj_b    [4096 x 3072]
__global__ void __launch_bounds__(256) k_qkv(const _Float16* __restrict__ A,
                                             const _Float16* __restrict__ W,
                                             const float* __restrict__ bias,
                                             _Float16* __restrict__ C) {
  gemm_core<2, 4>(A, DQ, W, DQ, DQ, [&](int m, int n, float v) {
    C[(long)m * 3072 + n] = (_Float16)(v + bias[n]);
  });
}

// edge_out = edge + (Q K^T)/8  (this IS the pre-softmax logit tensor t)
__global__ void __launch_bounds__(256) k_qk(const _Float16* __restrict__ qkv,
                                            const float* __restrict__ edge,
                                            float* __restrict__ eout) {
  const int z = blockIdx.z, b = z >> 4, hh = z & 15;
  const _Float16* q = qkv + b * 3072 + hh * HDQ;
  const _Float16* k = qkv + b * 3072 + DQ + hh * HDQ;
  const long base = (long)z << 20;
  gemm_core<2, 4>(q, (long)BQ * 3072, k, (long)BQ * 3072, HDQ,
                  [&](int m, int n, float v) {
    const long idx = base + ((long)m << 10) + n;
    eout[idx] = edge[idx] + v * 0.125f;   // 1/sqrt(HD)
  });
}

// out_heads = attn @ V (via V^T rows); merge heads on write: [L,B,H*HD]
__global__ void __launch_bounds__(256) k_av(const _Float16* __restrict__ attn,
                                            const _Float16* __restrict__ vt,
                                            _Float16* __restrict__ out) {
  const int z = blockIdx.z, b = z >> 4, hh = z & 15;
  gemm_core<4, 2>(attn + ((long)z << 20), LQ, vt + ((long)z << 16), LQ, LQ,
                  [&](int m, int n, float v) {
    out[((long)m * BQ + b) * DQ + hh * HDQ + n] = (_Float16)v;
  });
}

// x2 = x + out_heads @ out_proj_w^T + out_proj_b
__global__ void __launch_bounds__(256) k_outproj(const _Float16* __restrict__ A,
                                                 const _Float16* __restrict__ W,
                                                 const float* __restrict__ bias,
                                                 const float* __restrict__ xin,
                                                 float* __restrict__ x2) {
  gemm_core<2, 4>(A, DQ, W, DQ, DQ, [&](int m, int n, float v) {
    const long idx = (long)m * DQ + n;
    x2[idx] = xin[idx] + v + bias[n];
  });
}

// ff = gelu_exact(ln2(x2) @ lin1_w^T + lin1_b)   [4096 x 4096]
__global__ void __launch_bounds__(256) k_ffn1(const _Float16* __restrict__ A,
                                              const _Float16* __restrict__ W,
                                              const float* __restrict__ bias,
                                              _Float16* __restrict__ C) {
  gemm_core<2, 4>(A, DQ, W, DQ, DQ, [&](int m, int n, float v) {
    const float t = v + bias[n];
    const float g = 0.5f * t * (1.0f + erff(t * 0.70710678118654752f));
    C[(long)m * DFFQ + n] = (_Float16)g;
  });
}

// x_out = x2 + ff @ lin2_w^T + lin2_b
__global__ void __launch_bounds__(256) k_ffn2(const _Float16* __restrict__ A,
                                              const _Float16* __restrict__ W,
                                              const float* __restrict__ bias,
                                              const float* __restrict__ x2,
                                              float* __restrict__ xo) {
  gemm_core<2, 4>(A, DFFQ, W, DFFQ, DFFQ, [&](int m, int n, float v) {
    const long idx = (long)m * DQ + n;
    xo[idx] = x2[idx] + v + bias[n];
  });
}

// ---------------------------------------------------------------------------
extern "C" void kernel_launch(void* const* d_in, const int* in_sizes, int n_in,
                              void* d_out, int out_size, void* d_ws, size_t ws_size,
                              hipStream_t stream) {
  const float* x     = (const float*)d_in[0];
  const float* edge  = (const float*)d_in[1];
  const float* in_w  = (const float*)d_in[2];
  const float* in_b  = (const float*)d_in[3];
  const float* out_w = (const float*)d_in[4];
  const float* out_b = (const float*)d_in[5];
  const float* l1_w  = (const float*)d_in[6];
  const float* l1_b  = (const float*)d_in[7];
  const float* l2_w  = (const float*)d_in[8];
  const float* l2_b  = (const float*)d_in[9];
  const float* n1_w  = (const float*)d_in[10];
  const float* n1_b  = (const float*)d_in[11];
  const float* n2_w  = (const float*)d_in[12];
  const float* n2_b  = (const float*)d_in[13];

  float* x_out    = (float*)d_out;                  // [L,B,D]
  float* edge_out = x_out + (long)TOK * DQ;         // [B*H,L,L]

  char* wsp = (char*)d_ws;
  size_t off = 0;
  auto take = [&](size_t bytes) -> char* {
    char* p = wsp + off;
    off = (off + bytes + 255) & ~(size_t)255;
    return p;
  };
  _Float16* w_in_h  = (_Float16*)take((size_t)3072 * 1024 * 2);
  _Float16* w_out_h = (_Float16*)take((size_t)1024 * 1024 * 2);
  _Float16* w_l1_h  = (_Float16*)take((size_t)4096 * 1024 * 2);
  _Float16* w_l2_h  = (_Float16*)take((size_t)4096 * 1024 * 2);
  _Float16* h_h     = (_Float16*)take((size_t)TOK * 1024 * 2);
  _Float16* qkv_h   = (_Float16*)take((size_t)TOK * 3072 * 2);
  _Float16* vt_h    = (_Float16*)take((size_t)ZB * 64 * 1024 * 2);
  _Float16* attn_h  = (_Float16*)take((size_t)ZB * 1024 * 1024 * 2);
  _Float16* out_h   = (_Float16*)take((size_t)TOK * 1024 * 2);
  float*    x2      = (float*)take((size_t)TOK * 1024 * 4);
  _Float16* h2_h    = (_Float16*)take((size_t)TOK * 1024 * 2);
  _Float16* ff_h    = (_Float16*)take((size_t)TOK * 4096 * 2);

  // Weight conversion fp32 -> f16 (once per launch; deterministic)
  k_f2h<<<3072, 256, 0, stream>>>(in_w, w_in_h, 3072 * 1024);
  k_f2h<<<1024, 256, 0, stream>>>(out_w, w_out_h, 1024 * 1024);
  k_f2h<<<4096, 256, 0, stream>>>(l1_w, w_l1_h, 4096 * 1024);
  k_f2h<<<4096, 256, 0, stream>>>(l2_w, w_l2_h, 4096 * 1024);

  // ln1 -> QKV -> logits(edge_out) -> softmax -> attn@V -> out_proj(+res)
  k_ln<<<TOK, 256, 0, stream>>>(x, n1_w, n1_b, h_h);
  k_qkv<<<dim3(3072 / 128, TOK / 128, 1), 256, 0, stream>>>(h_h, w_in_h, in_b, qkv_h);
  k_qk<<<dim3(LQ / 128, LQ / 128, ZB), 256, 0, stream>>>(qkv_h, edge, edge_out);
  k_softmax<<<ZB * LQ, 256, 0, stream>>>(edge_out, attn_h);
  k_vtrans<<<dim3(LQ / 256, HDQ, ZB), 256, 0, stream>>>(qkv_h, vt_h);
  k_av<<<dim3(HDQ / 64, LQ / 256, ZB), 256, 0, stream>>>(attn_h, vt_h, out_h);
  k_outproj<<<dim3(DQ / 128, TOK / 128, 1), 256, 0, stream>>>(out_h, w_out_h, out_b, x, x2);

  // ln2 -> FFN (gelu fused) -> residual to x_out
  k_ln<<<TOK, 256, 0, stream>>>(x2, n2_w, n2_b, h2_h);
  k_ffn1<<<dim3(DFFQ / 128, TOK / 128, 1), 256, 0, stream>>>(h2_h, w_l1_h, l1_b, ff_h);
  k_ffn2<<<dim3(DQ / 128, TOK / 128, 1), 256, 0, stream>>>(ff_h, w_l2_h, l2_b, x2, x_out);
}
